// RNN_show_attend_tell_64123861729793
// MI455X (gfx1250) — matrix operand order repaired and checked by reading, not compiled, not run on hardware
//
#include <hip/hip_runtime.h>
#include <hip/hip_bf16.h>

// Model dims (fixed by reference)
#define B_ 128
#define P_ 196
#define E_ 2048
#define H_ 512
#define V_ 10000
#define T_ 20

typedef __bf16 bf16_t;
typedef __attribute__((ext_vector_type(16))) __bf16 v16bf;
typedef __attribute__((ext_vector_type(8)))  float   v8f;
typedef __attribute__((ext_vector_type(4)))  int     v4i;

union Frag16 { v16bf v; v4i q[2]; };   // 32 bytes: one WMMA A/B operand
union Bf8    { v4i   q; __bf16 h[8]; };// 16 bytes: 8 bf16

__device__ __forceinline__ float sigmoidf_(float x) { return 1.f / (1.f + __expf(-x)); }

// ---------------------------------------------------------------------------
// bf16 WMMA GEMM (guard-free hot loop): C = act( A[M,K] * Bt[N,K]^T + bias (+C) )
// A row-major bf16, Bt = B^T ([N,K] row-major bf16), C f32.
// Block: 256 threads = 8 waves arranged 2(M) x 4(N); wave tile 16 x 16*NF.
// Requires: grid.x * 64*NF == N region, grid.y * 32 == M, K % 32 == 0.
// Fragment layouts per CDNA5 ISA 7.12.2:
//   A 16x32 : lane m=lane&15; elems 0-7 -> K = k0+8*hlf+i ; 8-15 -> K+16
//   B 32x16 : lane n=lane&15; elems 0-7 -> K = k0+8*hlf+i ; 8-15 -> K+16
//   C 16x16 : lane n=lane&15; vgpr r    -> M = r + 8*hlf
// ---------------------------------------------------------------------------
template <int NF>
__global__ __launch_bounds__(256) void gemm_bf16_wmma(
    const bf16_t* __restrict__ A, int lda,
    const bf16_t* __restrict__ Bt, int ldb,
    float* __restrict__ C, long ldc,
    const float* __restrict__ bias,
    bf16_t* __restrict__ Cbf, int ldcb,
    int K, int act, int accumulate) {
  const int tid  = threadIdx.x;
  const int lane = tid & 31;
  const int hlf  = lane >> 4;
  const int lm   = lane & 15;
  // wave-uniform scalars (readfirstlane => provably scalar, no EXEC games)
  const int wid  = __builtin_amdgcn_readfirstlane(tid >> 5);
  const int wm   = wid >> 2;                       // 0..1
  const int wn   = wid & 3;                        // 0..3
  const int m0    = blockIdx.y * 32 + wm * 16;
  const int nbase = blockIdx.x * (64 * NF) + wn * (16 * NF);

  v8f acc[NF];
#pragma unroll
  for (int j = 0; j < NF; ++j) acc[j] = (v8f){0.f, 0.f, 0.f, 0.f, 0.f, 0.f, 0.f, 0.f};

  const bf16_t* Arow = A + (size_t)(m0 + lm) * lda + hlf * 8;
  const bf16_t* Brow0 = Bt + (size_t)(nbase + lm) * ldb + hlf * 8;

  for (int k0 = 0; k0 < K; k0 += 32) {
    Frag16 a;
    a.q[0] = *(const v4i*)(Arow + k0);
    a.q[1] = *(const v4i*)(Arow + k0 + 16);
#pragma unroll
    for (int j = 0; j < NF; ++j) {
      const bf16_t* Brow = Brow0 + (size_t)(j * 16) * ldb + k0;
      Frag16 b;
      b.q[0] = *(const v4i*)(Brow);
      b.q[1] = *(const v4i*)(Brow + 16);
      acc[j] = __builtin_amdgcn_wmma_f32_16x16x32_bf16(
          false, a.v, false, b.v, (short)0, acc[j], false, false);
    }
  }

#pragma unroll
  for (int j = 0; j < NF; ++j) {
    const int n = nbase + j * 16 + lm;
    const float bv = bias ? bias[n] : 0.f;
#pragma unroll
    for (int r = 0; r < 8; ++r) {
      const int m = m0 + r + hlf * 8;
      float v = acc[j][r] + bv;
      if (accumulate) v += C[(size_t)m * ldc + n];
      if (act == 1) v = tanhf(v);
      else if (act == 2) v = sigmoidf_(v);
      C[(size_t)m * ldc + n] = v;
      if (Cbf) Cbf[(size_t)m * ldcb + n] = (bf16_t)v;
    }
  }
}

// Tail kernel: one 16-wide column strip at n0; 8 waves stacked along M.
// Requires M % 128 == 0 (used only for the V=10000 out-GEMM tail, M=128).
__global__ __launch_bounds__(256) void gemm_col16_wmma(
    const bf16_t* __restrict__ A, int lda,
    const bf16_t* __restrict__ Bt, int ldb,
    float* __restrict__ C, long ldc,
    const float* __restrict__ bias,
    bf16_t* __restrict__ Cbf, int ldcb,
    int n0, int K, int act, int accumulate) {
  const int tid  = threadIdx.x;
  const int lane = tid & 31;
  const int hlf  = lane >> 4;
  const int lm   = lane & 15;
  const int wid  = __builtin_amdgcn_readfirstlane(tid >> 5);
  const int m0   = blockIdx.x * 128 + wid * 16;

  v8f acc = (v8f){0.f, 0.f, 0.f, 0.f, 0.f, 0.f, 0.f, 0.f};
  const bf16_t* Arow = A + (size_t)(m0 + lm) * lda + hlf * 8;
  const bf16_t* Brow = Bt + (size_t)(n0 + lm) * ldb + hlf * 8;

  for (int k0 = 0; k0 < K; k0 += 32) {
    Frag16 a, b;
    a.q[0] = *(const v4i*)(Arow + k0);
    a.q[1] = *(const v4i*)(Arow + k0 + 16);
    b.q[0] = *(const v4i*)(Brow + k0);
    b.q[1] = *(const v4i*)(Brow + k0 + 16);
    acc = __builtin_amdgcn_wmma_f32_16x16x32_bf16(
        false, a.v, false, b.v, (short)0, acc, false, false);
  }

  const int n = n0 + lm;
  const float bv = bias ? bias[n] : 0.f;
#pragma unroll
  for (int r = 0; r < 8; ++r) {
    const int m = m0 + r + hlf * 8;
    float v = acc[r] + bv;
    if (accumulate) v += C[(size_t)m * ldc + n];
    if (act == 1) v = tanhf(v);
    else if (act == 2) v = sigmoidf_(v);
    C[(size_t)m * ldc + n] = v;
    if (Cbf) Cbf[(size_t)m * ldcb + n] = (bf16_t)v;
  }
}

// ---------------------------------------------------------------------------
// Prep kernels
// ---------------------------------------------------------------------------
__global__ void cvt_bf16_kernel(const float* __restrict__ src, bf16_t* __restrict__ dst, size_t n) {
  size_t i = (size_t)blockIdx.x * 256 + threadIdx.x;
  if (i < n) dst[i] = (bf16_t)src[i];
}

__global__ void transpose_bf16_kernel(const float* __restrict__ src, bf16_t* __restrict__ dst,
                                      int K, int N) {
  size_t idx = (size_t)blockIdx.x * 256 + threadIdx.x;
  if (idx >= (size_t)K * N) return;
  int k = (int)(idx / N), n = (int)(idx % N);
  dst[(size_t)n * K + k] = (bf16_t)src[idx];
}

__global__ void mean_img_kernel(const float* __restrict__ img, bf16_t* __restrict__ avg_bf) {
  int idx = blockIdx.x * 256 + threadIdx.x;          // B_*E_ exact
  int b = idx / E_, e = idx % E_;
  const float* base = img + (size_t)b * P_ * E_ + e;
  float s = 0.f;
  for (int p = 0; p < P_; ++p) s += base[(size_t)p * E_];
  avg_bf[idx] = (bf16_t)(s * (1.f / P_));
}

// ---------------------------------------------------------------------------
// Attention scores: e[b,p] = sum_h v[h]*tanh(Ws[b,p,h] + hU[b,h]) + v_b
// One block per b; 8 waves sweep p; 32 lanes x 16 elems reduce H=512.
// ---------------------------------------------------------------------------
__global__ void attn_scores_kernel(const float* __restrict__ Ws, const float* __restrict__ hU,
                                   const float* __restrict__ v_w, const float* __restrict__ v_b,
                                   float* __restrict__ e) {
  __shared__ float sh_hU[H_];
  __shared__ float sh_v[H_];
  const int b = blockIdx.x;
  for (int i = threadIdx.x; i < H_; i += blockDim.x) {
    sh_hU[i] = hU[b * H_ + i];
    sh_v[i]  = v_w[i];
  }
  __syncthreads();
  const int wid = threadIdx.x >> 5, lane = threadIdx.x & 31;
  const float vb = v_b[0];
  for (int p = wid; p < P_; p += 8) {
    const float* row = Ws + ((size_t)b * P_ + p) * H_;
    float s = 0.f;
#pragma unroll 4
    for (int i = 0; i < 16; ++i) {
      int h = lane * 16 + i;
      s += sh_v[h] * tanhf(row[h] + sh_hU[h]);
    }
    for (int off = 16; off > 0; off >>= 1) s += __shfl_xor(s, off, 32);
    if (lane == 0) e[b * P_ + p] = s + vb;
  }
}

// ---------------------------------------------------------------------------
// Softmax over P + context = alpha . img  (img kept bf16 -> L2-resident)
// One block (256 thr) per b; each thread owns 8 consecutive e-channels (16B loads).
// ---------------------------------------------------------------------------
__global__ void softmax_context_kernel(const float* __restrict__ e, const bf16_t* __restrict__ img,
                                       float* __restrict__ ctx, float* __restrict__ alphas, int t) {
  __shared__ float red[256];
  __shared__ float al[P_];
  const int b = blockIdx.x, tid = threadIdx.x;
  float x = (tid < P_) ? e[b * P_ + tid] : -3.4e38f;
  red[tid] = x; __syncthreads();
  for (int s = 128; s > 0; s >>= 1) { if (tid < s) red[tid] = fmaxf(red[tid], red[tid + s]); __syncthreads(); }
  const float mx = red[0]; __syncthreads();
  float ex = (tid < P_) ? __expf(x - mx) : 0.f;
  red[tid] = ex; __syncthreads();
  for (int s = 128; s > 0; s >>= 1) { if (tid < s) red[tid] += red[tid + s]; __syncthreads(); }
  const float inv = 1.f / red[0];
  __syncthreads();
  if (tid < P_) {
    float a = ex * inv;
    al[tid] = a;
    alphas[((size_t)b * T_ + t) * P_ + tid] = a;
  }
  __syncthreads();

  float acc[8] = {0, 0, 0, 0, 0, 0, 0, 0};
  const int e0 = tid * 8;                               // 256*8 == E_
  const bf16_t* base = img + (size_t)b * P_ * E_ + e0;
  for (int p = 0; p < P_; ++p) {
    Bf8 u; u.q = *(const v4i*)(base + (size_t)p * E_);
    const float a = al[p];
#pragma unroll
    for (int i = 0; i < 8; ++i) acc[i] += a * (float)u.h[i];
  }
  float* cb = ctx + (size_t)b * E_ + e0;
#pragma unroll
  for (int i = 0; i < 8; ++i) cb[i] = acc[i];
}

// x = [emb0 (H) | gate*context (E)] as bf16, per batch row
__global__ void build_x_kernel(const bf16_t* __restrict__ emb0, const float* __restrict__ gate,
                               const float* __restrict__ ctx, bf16_t* __restrict__ x) {
  int idx = blockIdx.x * 256 + threadIdx.x;            // B_*(H_+E_) exact
  int b = idx / (H_ + E_), k = idx % (H_ + E_);
  if (k < H_) x[idx] = emb0[k];
  else {
    size_t o = (size_t)b * E_ + (k - H_);
    x[idx] = (bf16_t)(gate[o] * ctx[o]);
  }
}

// LSTM cell pointwise (gate order i,f,g,o)
__global__ void lstm_pointwise_kernel(const float* __restrict__ gates, float* __restrict__ c,
                                      float* __restrict__ h, bf16_t* __restrict__ h_bf) {
  int idx = blockIdx.x * 256 + threadIdx.x;            // B_*H_ exact
  int b = idx / H_, j = idx % H_;
  const float* g = gates + (size_t)b * 4 * H_;
  float ig = sigmoidf_(g[j]);
  float fg = sigmoidf_(g[H_ + j]);
  float gg = tanhf(g[2 * H_ + j]);
  float og = sigmoidf_(g[3 * H_ + j]);
  float cn = fg * c[idx] + ig * gg;
  c[idx] = cn;
  float hn = og * tanhf(cn);
  h[idx] = hn;
  h_bf[idx] = (bf16_t)hn;
}

// ---------------------------------------------------------------------------
extern "C" void kernel_launch(void* const* d_in, const int* in_sizes, int n_in,
                              void* d_out, int out_size, void* d_ws, size_t ws_size,
                              hipStream_t stream) {
  (void)in_sizes; (void)n_in; (void)out_size; (void)ws_size;
  const float* img   = (const float*)d_in[0];
  /* d_in[1] captions: unused — training path feeds emb[0] each step */
  const float* U_w   = (const float*)d_in[2];
  const float* U_b   = (const float*)d_in[3];
  const float* W_w   = (const float*)d_in[4];
  const float* W_b   = (const float*)d_in[5];
  const float* v_w   = (const float*)d_in[6];
  const float* v_b   = (const float*)d_in[7];
  const float* ih_w  = (const float*)d_in[8];
  const float* ih_b  = (const float*)d_in[9];
  const float* ic_w  = (const float*)d_in[10];
  const float* ic_b  = (const float*)d_in[11];
  const float* fb_w  = (const float*)d_in[12];
  const float* fb_b  = (const float*)d_in[13];
  const float* out_w = (const float*)d_in[14];
  const float* out_b = (const float*)d_in[15];
  const float* emb   = (const float*)d_in[16];
  const float* Wih   = (const float*)d_in[17];
  const float* Whh   = (const float*)d_in[18];
  const float* bih   = (const float*)d_in[19];
  const float* bhh   = (const float*)d_in[20];

  float* preds  = (float*)d_out;                       // [B,T,V]
  float* alphas = preds + (size_t)B_ * T_ * V_;        // [B,T,P]

  // --- bump allocator over workspace (256B aligned) ---
  char* wsp = (char*)d_ws;
  size_t off = 0;
  auto alloc = [&](size_t bytes) -> void* {
    off = (off + 255) & ~(size_t)255;
    void* p = wsp + off;
    off += bytes;
    return p;
  };
  bf16_t* img_bf  = (bf16_t*)alloc((size_t)B_ * P_ * E_ * 2);   // 98 MB  (L2-resident)
  float*  Ws      = (float*) alloc((size_t)B_ * P_ * H_ * 4);   // 51 MB  (L2-resident)
  bf16_t* Ut      = (bf16_t*)alloc((size_t)H_ * H_ * 2);        // [H,H]
  bf16_t* WwT     = (bf16_t*)alloc((size_t)H_ * E_ * 2);        // [H,E]
  bf16_t* ihT     = (bf16_t*)alloc((size_t)H_ * E_ * 2);
  bf16_t* icT     = (bf16_t*)alloc((size_t)H_ * E_ * 2);
  bf16_t* fbT     = (bf16_t*)alloc((size_t)E_ * H_ * 2);        // [E,H]
  bf16_t* outT    = (bf16_t*)alloc((size_t)V_ * H_ * 2);        // [V,H]
  bf16_t* WihT    = (bf16_t*)alloc((size_t)4 * H_ * (H_ + E_) * 2); // [4H,2560]
  bf16_t* WhhT    = (bf16_t*)alloc((size_t)4 * H_ * H_ * 2);    // [4H,H]
  bf16_t* avg_bf  = (bf16_t*)alloc((size_t)B_ * E_ * 2);
  bf16_t* emb0_bf = (bf16_t*)alloc((size_t)H_ * 2);
  float*  h_f     = (float*) alloc((size_t)B_ * H_ * 4);
  float*  c_f     = (float*) alloc((size_t)B_ * H_ * 4);
  bf16_t* h_bf    = (bf16_t*)alloc((size_t)B_ * H_ * 2);
  float*  hU      = (float*) alloc((size_t)B_ * H_ * 4);
  float*  e_sc    = (float*) alloc((size_t)B_ * P_ * 4);
  float*  ctx     = (float*) alloc((size_t)B_ * E_ * 4);
  float*  gate    = (float*) alloc((size_t)B_ * E_ * 4);
  bf16_t* x_bf    = (bf16_t*)alloc((size_t)B_ * (H_ + E_) * 2);
  float*  gates   = (float*) alloc((size_t)B_ * 4 * H_ * 4);

  auto gemm = [&](const bf16_t* A, int lda, const bf16_t* Bt, int ldb, float* C, long ldc,
                  const float* bias, bf16_t* Cbf, int ldcb, int M, int N, int K, int act, int acc) {
    const int N4 = (N / 256) * 256;
    if (N4 > 0) {
      dim3 grid(N4 / 256, M / 32);
      gemm_bf16_wmma<4><<<grid, 256, 0, stream>>>(A, lda, Bt, ldb, C, ldc, bias, Cbf, ldcb,
                                                  K, act, acc);
    }
    for (int n0 = N4; n0 < N; n0 += 16) {          // only hits for V=10000 (one strip)
      gemm_col16_wmma<<<M / 128, 256, 0, stream>>>(A, lda, Bt, ldb, C, ldc, bias, Cbf, ldcb,
                                                   n0, K, act, acc);
    }
  };
  auto nb = [](size_t n) { return (unsigned)((n + 255) / 256); };

  // ---- prep: bf16 conversions + weight transposes to [N,K] ----
  cvt_bf16_kernel<<<nb((size_t)B_ * P_ * E_), 256, 0, stream>>>(img, img_bf, (size_t)B_ * P_ * E_);
  transpose_bf16_kernel<<<nb((size_t)H_ * H_), 256, 0, stream>>>(U_w, Ut, H_, H_);
  transpose_bf16_kernel<<<nb((size_t)E_ * H_), 256, 0, stream>>>(W_w, WwT, E_, H_);
  transpose_bf16_kernel<<<nb((size_t)E_ * H_), 256, 0, stream>>>(ih_w, ihT, E_, H_);
  transpose_bf16_kernel<<<nb((size_t)E_ * H_), 256, 0, stream>>>(ic_w, icT, E_, H_);
  transpose_bf16_kernel<<<nb((size_t)H_ * E_), 256, 0, stream>>>(fb_w, fbT, H_, E_);
  transpose_bf16_kernel<<<nb((size_t)H_ * V_), 256, 0, stream>>>(out_w, outT, H_, V_);
  transpose_bf16_kernel<<<nb((size_t)(H_ + E_) * 4 * H_), 256, 0, stream>>>(Wih, WihT, H_ + E_, 4 * H_);
  transpose_bf16_kernel<<<nb((size_t)H_ * 4 * H_), 256, 0, stream>>>(Whh, WhhT, H_, 4 * H_);
  cvt_bf16_kernel<<<nb(H_), 256, 0, stream>>>(emb, emb0_bf, (size_t)H_);  // emb row 0
  mean_img_kernel<<<(B_ * E_) / 256, 256, 0, stream>>>(img, avg_bf);

  // h0 = tanh(avg@init_h_w + b)  (also emit bf16);  c0 = tanh(avg@init_c_w + b)
  gemm(avg_bf, E_, ihT, E_, h_f, H_, ih_b, h_bf, H_, B_, H_, E_, /*tanh*/1, 0);
  gemm(avg_bf, E_, icT, E_, c_f, H_, ic_b, nullptr, 0, B_, H_, E_, /*tanh*/1, 0);

  // W_s = img@W_w + b   [25088,2048]x[2048,512] — the big one-time GEMM
  gemm(img_bf, E_, WwT, E_, Ws, H_, W_b, nullptr, 0, B_ * P_, H_, E_, 0, 0);

  // ---- T sequential decode steps ----
  for (int t = 0; t < T_; ++t) {
    // hU = h@U_w + U_b
    gemm(h_bf, H_, Ut, H_, hU, H_, U_b, nullptr, 0, B_, H_, H_, 0, 0);
    attn_scores_kernel<<<B_, 256, 0, stream>>>(Ws, hU, v_w, v_b, e_sc);
    softmax_context_kernel<<<B_, 256, 0, stream>>>(e_sc, img_bf, ctx, alphas, t);
    // gate = sigmoid(h@f_beta_w + b)
    gemm(h_bf, H_, fbT, H_, gate, E_, fb_b, nullptr, 0, B_, E_, H_, /*sigmoid*/2, 0);
    build_x_kernel<<<(B_ * (H_ + E_)) / 256, 256, 0, stream>>>(emb0_bf, gate, ctx, x_bf);
    // gates = x@Wih + bih, then += h@Whh + bhh
    gemm(x_bf, H_ + E_, WihT, H_ + E_, gates, 4 * H_, bih, nullptr, 0, B_, 4 * H_, H_ + E_, 0, 0);
    gemm(h_bf, H_, WhhT, H_, gates, 4 * H_, bhh, nullptr, 0, B_, 4 * H_, H_, 0, 1);
    lstm_pointwise_kernel<<<(B_ * H_) / 256, 256, 0, stream>>>(gates, c_f, h_f, h_bf);
    // preds[:,t,:] = h@out_w + out_b   (write straight into d_out, row stride T*V)
    gemm(h_bf, H_, outT, H_, preds + (size_t)t * V_, (long)T_ * V_, out_b, nullptr, 0,
         B_, V_, H_, 0, 0);
  }
}
